// decode_param_CNN_Fourier_85650237817231
// MI455X (gfx1250) — matrix operand order repaired and checked by reading, not compiled, not run on hardware
//
#include <hip/hip_runtime.h>
#include <hip/hip_bf16.h>

typedef __attribute__((ext_vector_type(16))) __bf16 v16bf;
typedef __attribute__((ext_vector_type(8)))  float  v8f;

#define NB      32
#define NT      64
#define KDIM    80          // 40 kl-pairs x (cos,sin)
#define HW      10000
#define PTILES  625         // 10000 / 16
#define TTILES  4           // 64 / 16
#define NPARAMS 5376
#define BIAS1   330         // kl[-1,0]*(2*39+4)+2 = 4*82+2

#define A_ELEMS (PTILES * 3 * 512)          // 960000 bf16
#define B_ELEMS (NB * 2 * TTILES * 3 * 512) // 393216 bf16

// kl table row q: q<4 -> (0, q+1); else qq=q-4 -> (1+qq/9, qq%9-4)
static __device__ __forceinline__ void kl_of_q(int q, int& kl0, int& kl1) {
    if (q < 4) { kl0 = 0; kl1 = q + 1; }
    else { int qq = q - 4; kl0 = 1 + qq / 9; kl1 = (qq % 9) - 4; }
}

// ST[t,q]: q==0 -> t*4+4 ; q>=1 -> kl[q-1,0]*(2q+4)+4 (time-independent)
static __device__ __forceinline__ int st_of(int t, int q) {
    if (q == 0) return t * 4 + 4;
    int n = (q <= 4) ? 0 : (1 + (q - 5) / 9);
    return n * (2 * q + 4) + 4;
}

// Basis table in WMMA bf16 A-fragment layout.
// Layout: Atab[((pt*3 + kc)*32 + lane)*16 + e]
// A 16x32 bf16 fragment: lane<16: M=lane, K = e<8 ? e : e+8
//                        lane>=16: M=lane-16, K = e<8 ? e+8 : e+16
__global__ void build_atab_kernel(__bf16* __restrict__ Atab) {
    int idx = blockIdx.x * blockDim.x + threadIdx.x;
    if (idx >= A_ELEMS) return;
    int e  = idx & 15;
    int l  = (idx >> 4) & 31;
    int g  = idx >> 9;
    int kc = g % 3;
    int pt = g / 3;
    int M  = l & 15;
    int klocal = (l < 16) ? ((e < 8) ? e : e + 8)
                          : ((e < 8) ? e + 8 : e + 16);
    int kk = kc * 32 + klocal;
    float val = 0.0f;
    if (kk < KDIM) {
        int q = kk >> 1;
        int kl0, kl1; kl_of_q(q, kl0, kl1);
        int p  = pt * 16 + M;
        int gi = p / 100, gj = p % 100;
        float ph = 6.28318530717958647692f * 0.01f * (float)(kl0 * gi + kl1 * gj);
        val = (kk & 1) ? sinf(ph) : cosf(ph);
    }
    Atab[idx] = (__bf16)val;
}

// Gathered coefficients in WMMA bf16 B-fragment layout.
// Layout: Bcoef[(((bm*4 + tt)*3 + kc)*32 + lane)*16 + e]
// B 32x16 bf16 fragment: lane<16: N=lane, K = e<8 ? e : e+8 ; lane>=16: N=lane-16, K = e<8 ? e+8 : e+16
__global__ void build_bcoef_kernel(const float* __restrict__ params,
                                   __bf16* __restrict__ Bcoef) {
    int idx = blockIdx.x * blockDim.x + threadIdx.x;
    if (idx >= B_ELEMS) return;
    int e  = idx & 15;
    int l  = (idx >> 4) & 31;
    int g  = idx >> 9;         // (bm*4+tt)*3 + kc
    int kc = g % 3;
    int h  = g / 3;
    int tt = h & 3;
    int bm = h >> 2;           // 0..63
    int m  = bm & 1;
    int b  = bm >> 1;
    int N  = l & 15;
    int klocal = (l < 16) ? ((e < 8) ? e : e + 8) : ((e < 8) ? e + 8 : e + 16);
    int kk = kc * 32 + klocal;
    float val = 0.0f;
    if (kk < KDIM) {
        int q   = kk >> 1;
        int t   = tt * 16 + N;
        int col = st_of(t, q) + (kk & 1) + 2 * m;   // ST+0/1 (m=0), ST+2/3 (m=1)
        val = params[(size_t)b * NPARAMS + col];
    }
    Bcoef[idx] = (__bf16)val;
}

// One wave per (bm, pt) strip: A fragments loaded once, swept across the
// 4 t-tiles (12 WMMAs / wave). K = 96 (80 zero-padded), 3 chunks of 32.
__global__ void __launch_bounds__(256)
fourier_wmma_kernel(const float* __restrict__ params,
                    const __bf16* __restrict__ Atab,
                    const __bf16* __restrict__ Bcoef,
                    float* __restrict__ out) {
    int wid  = blockIdx.x * (blockDim.x >> 5) + (threadIdx.x >> 5); // 40,000
    int lane = threadIdx.x & 31;
    int pt   = wid % PTILES;
    int bm   = wid / PTILES;           // b*2 + m
    int m    = bm & 1;
    int b    = bm >> 1;

    float bias = params[(size_t)b * NPARAMS + BIAS1 + m];

    // Basis fragments for this p-tile: resident in registers for all 4 t-tiles
    const v16bf* Af = (const v16bf*)Atab + (size_t)pt * 3 * 32;
    v16bf a0 = Af[0 * 32 + lane];
    v16bf a1 = Af[1 * 32 + lane];
    v16bf a2 = Af[2 * 32 + lane];

    const v16bf* Bf = (const v16bf*)Bcoef + (size_t)bm * TTILES * 3 * 32;

    int prow0 = pt * 16 + ((lane >> 4) << 3);        // D rows: r (+8 for hi half)
    float* obase = out + ((size_t)bm * HW + prow0) * NT + (lane & 15);

#pragma unroll
    for (int tt = 0; tt < TTILES; ++tt) {
        v16bf b0 = Bf[(tt * 3 + 0) * 32 + lane];
        v16bf b1 = Bf[(tt * 3 + 1) * 32 + lane];
        v16bf b2 = Bf[(tt * 3 + 2) * 32 + lane];

        v8f acc;
#pragma unroll
        for (int r = 0; r < 8; ++r) acc[r] = bias;

        acc = __builtin_amdgcn_wmma_f32_16x16x32_bf16(false, a0, false, b0,
                                                      (short)0, acc, false, false);
        acc = __builtin_amdgcn_wmma_f32_16x16x32_bf16(false, a1, false, b1,
                                                      (short)0, acc, false, false);
        acc = __builtin_amdgcn_wmma_f32_16x16x32_bf16(false, a2, false, b2,
                                                      (short)0, acc, false, false);

        // D layout: lanes 0-15: (M=r, N=lane); lanes 16-31: (M=r+8, N=lane-16)
        float* o = obase + tt * 16;
#pragma unroll
        for (int r = 0; r < 8; ++r)
            o[(size_t)r * NT] = acc[r];
    }
}

extern "C" void kernel_launch(void* const* d_in, const int* in_sizes, int n_in,
                              void* d_out, int out_size, void* d_ws, size_t ws_size,
                              hipStream_t stream) {
    (void)in_sizes; (void)n_in; (void)out_size; (void)ws_size;
    const float* params = (const float*)d_in[0];
    float* out = (float*)d_out;

    __bf16* Atab  = (__bf16*)d_ws;                                  // 1,920,000 B
    __bf16* Bcoef = (__bf16*)((char*)d_ws + (size_t)A_ELEMS * 2);   //   786,432 B

    build_atab_kernel<<<(A_ELEMS + 255) / 256, 256, 0, stream>>>(Atab);
    build_bcoef_kernel<<<(B_ELEMS + 255) / 256, 256, 0, stream>>>(params, Bcoef);

    int total_waves = NB * 2 * PTILES;            // 40,000 strips, 1 wave each
    fourier_wmma_kernel<<<total_waves / 8, 256, 0, stream>>>(params, Atab, Bcoef, out);
}
